// HGAT_56951266345673
// MI455X (gfx1250) — matrix operand order repaired and checked by reading, not compiled, not run on hardware
//
#include <hip/hip_runtime.h>
#include <hip/hip_bf16.h>

// ---------------------------------------------------------------------------
// HGAT on MI455X (gfx1250, wave32, WMMA + async-to-LDS).
// All heavy GEMMs use v_wmma_f32_16x16x32_f16 (f16 in, fp32 accumulate).
// Attention matrix is never materialized: softmax stats are precomputed
// (logits are rank-1: s_src[i]+s_dst[j] through leakyrelu+mask), and the
// A-operand of the att@h GEMM is regenerated per 16x32 tile from LDS.
// Cooperative tile staging uses GLOBAL_LOAD_ASYNC_TO_LDS_B128 with a
// triple-buffered pipeline (ASYNCcnt depth 1) so the next tile's fetch
// overlaps the current tile's WMMAs; one s_barrier per K-step.
// ---------------------------------------------------------------------------

typedef __attribute__((ext_vector_type(16))) _Float16 v16h;
typedef __attribute__((ext_vector_type(8)))  _Float16 v8h;
typedef __attribute__((ext_vector_type(8)))  float    v8f;

constexpr int N_     = 2048;
constexpr int NFEAT  = 512;
constexpr int NHID   = 128;
constexpr int NHEADS = 4;
constexpr int P_     = 3;
constexpr int SHID   = 128;
constexpr int PH     = P_ * NHEADS;          // 12
constexpr int KF     = NHEADS * NHID;        // 512 (= m row width)
constexpr float ALPHA = 0.2f;

// ---------------------------------------------------------------------------
// WMMA helper: D = A(16x32 f16) * B(32x16 f16) + C(16x16 f32)
// ---------------------------------------------------------------------------
__device__ inline v8f wmma_f16(v16h a, v16h b, v8f c) {
  return __builtin_amdgcn_wmma_f32_16x16x32_f16(
      /*neg_a=*/false, a, /*neg_b=*/false, b,
      /*c_mod=*/(short)0, c, /*reuse_a=*/false, /*reuse_b=*/false);
}

// Async 16-byte global -> LDS copy, per-lane addresses (ISA 15.18 op 98).
// LDS address is taken from the vdst VGPR; generic-pointer low 32 bits == LDS offset.
__device__ __forceinline__ void async_ld_b128(unsigned lds_off,
                                              const void* gptr) {
  asm volatile("global_load_async_to_lds_b128 %0, %1, off"
               :
               : "v"(lds_off), "v"((unsigned long long)(uintptr_t)gptr)
               : "memory");
}
__device__ __forceinline__ void wait_async1() {
  asm volatile("s_wait_asynccnt 0x1" ::: "memory");
}
__device__ __forceinline__ void wait_async0() {
  asm volatile("s_wait_asynccnt 0x0" ::: "memory");
}
__device__ __forceinline__ unsigned lds_off_u32(const void* p) {
  return (unsigned)(uintptr_t)p;   // AS3 aperture: addr[31:0] is the offset
}

// A-operand per-lane layout (16-bit A 16x32, contiguous K like B):
//   lane L: row M = L&15 ; halves[0..7]=K kc..kc+7 ; halves[8..15]=K kc+8..kc+15
//   with kc = (L>>4)*16.
// B-operand per-lane layout (contiguous-chunk analogy of sparse-B table):
//   lane L: col N = L&15 ; halves[0..15] = K (L>>4)*16 .. +15.

// ---------------------------------------------------------------------------
// convert kernels
// ---------------------------------------------------------------------------
__global__ void cvt_f32_to_f16_kernel(const float* __restrict__ src,
                                      _Float16* __restrict__ dst) {
  int i = blockIdx.x * blockDim.x + threadIdx.x;
  dst[i] = (_Float16)src[i];
}

// W_node (PH, NFEAT, NHID) fp32 -> WT (PH, NHID, NFEAT) f16
__global__ void wnodeT_kernel(const float* __restrict__ W,
                              _Float16* __restrict__ WT) {
  int i = blockIdx.x * blockDim.x + threadIdx.x;     // PH*NHID*NFEAT
  int f  = i % NFEAT;
  int r  = i / NFEAT;
  int d  = r % NHID;
  int ph = r / NHID;
  WT[i] = (_Float16)W[((size_t)ph * NFEAT + f) * NHID + d];
}

// W_sem (KF, SHID) fp32 -> WsemT (SHID, KF) f16
__global__ void wsemT_kernel(const float* __restrict__ W,
                             _Float16* __restrict__ WT) {
  int i = blockIdx.x * blockDim.x + threadIdx.x;     // SHID*KF
  int f = i % KF;
  int s = i / KF;
  WT[i] = (_Float16)W[(size_t)f * SHID + s];
}

// ---------------------------------------------------------------------------
// GEMM 1: h[ph] = X(2048x512) @ W[ph](512x128); store transposed f16 hT(ph,d,n)
// 8 waves/block, each wave: one 16-row x 128-col tile. grid = PH * (N/128).
// B tiles (identical for all 8 waves) are staged in LDS by async-to-LDS with
// a 3-deep pipeline: fetch of tile kt+1 overlaps WMMAs of tile kt.
// ---------------------------------------------------------------------------
__global__ __launch_bounds__(256) void gemm_h_kernel(
    const _Float16* __restrict__ xh, const _Float16* __restrict__ WT,
    _Float16* __restrict__ hT) {
  __shared__ __align__(32) _Float16 bS[3][NHID * 32];   // 3 x 8KB
  const int tid  = threadIdx.x;
  const int wave = tid >> 5;
  const int lane = tid & 31;
  const int ph      = blockIdx.x / (N_ / 128);
  const int rowblk  = blockIdx.x % (N_ / 128);
  const int rowbase = rowblk * 128 + wave * 16;
  const int mrow    = rowbase + (lane & 15);
  const int kc      = (lane >> 4) * 16;

  // async B loader mapping: 256 threads -> 128 d-rows x 2 half-chunks(16h=32B)
  const int ld_d = tid >> 1;
  const int ld_c = (tid & 1) * 16;
  const _Float16* gb = WT + ((size_t)ph * NHID + ld_d) * NFEAT + ld_c;

  v8f acc[8];
#pragma unroll
  for (int dt = 0; dt < 8; ++dt)
#pragma unroll
    for (int z = 0; z < 8; ++z) acc[dt][z] = 0.f;

  constexpr int KT = NFEAT / 32;    // 16 K-steps
  async_ld_b128(lds_off_u32(&bS[0][ld_d * 32 + ld_c]), gb);   // tile 0

  const _Float16* arow = xh + (size_t)mrow * NFEAT;
  for (int kt = 0; kt < KT; ++kt) {
    const int k0 = kt * 32;
    v16h A;
    {
      const v8h a0 = *(const v8h*)(arow + k0 + kc);
      const v8h a1 = *(const v8h*)(arow + k0 + kc + 8);
#pragma unroll
      for (int t = 0; t < 8; ++t) { A[t] = a0[t]; A[8 + t] = a1[t]; }
    }
    if (kt + 1 < KT) {
      async_ld_b128(lds_off_u32(&bS[(kt + 1) % 3][ld_d * 32 + ld_c]),
                    gb + (kt + 1) * 32);
      wait_async1();                 // tile kt complete, kt+1 in flight
    } else {
      wait_async0();
    }
    __syncthreads();
    const _Float16* bbuf = &bS[kt % 3][0];
    v16h Bt[8];
#pragma unroll
    for (int dt = 0; dt < 8; ++dt)
      Bt[dt] = *(const v16h*)(bbuf + (dt * 16 + (lane & 15)) * 32 +
                              (lane >> 4) * 16);
#pragma unroll
    for (int dt = 0; dt < 8; ++dt) acc[dt] = wmma_f16(A, Bt[dt], acc[dt]);
  }
  // D layout: lanes 0-15 col n=lane, M=r ; lanes 16-31 col n=lane-16, M=8+r
  const int nodebase = rowbase + 8 * (lane >> 4);
#pragma unroll
  for (int dt = 0; dt < 8; ++dt) {
    const int d = dt * 16 + (lane & 15);
    _Float16* orow = hT + ((size_t)ph * NHID + d) * N_;
#pragma unroll
    for (int r = 0; r < 8; ++r) orow[nodebase + r] = (_Float16)acc[dt][r];
  }
}

// ---------------------------------------------------------------------------
// per-node attention scalars: s_src = h.a_src, s_dst = h.a_dst
// ---------------------------------------------------------------------------
__global__ void node_scores_kernel(const _Float16* __restrict__ hT,
                                   const float* __restrict__ a_node,
                                   float* __restrict__ s_src,
                                   float* __restrict__ s_dst) {
  int i  = blockIdx.x * blockDim.x + threadIdx.x;    // PH*N_
  int n  = i % N_;
  int ph = i / N_;
  const float* av = a_node + (size_t)ph * 2 * NHID;
  float ss = 0.f, sd = 0.f;
  for (int d = 0; d < NHID; ++d) {
    float hv = (float)hT[((size_t)ph * NHID + d) * N_ + n];
    ss += hv * av[d];
    sd += hv * av[NHID + d];
  }
  s_src[i] = ss;
  s_dst[i] = sd;
}

// ---------------------------------------------------------------------------
// softmax stats: one wave per (p,row), all 4 heads per adj-row read.
// Online (max,sum) per head, then lane-shuffle merge.
// ---------------------------------------------------------------------------
__global__ __launch_bounds__(256) void stats_kernel(
    const float* __restrict__ adjs, const float* __restrict__ s_src,
    const float* __restrict__ s_dst, float* __restrict__ rmax,
    float* __restrict__ rinv) {
  const int wid  = (blockIdx.x * blockDim.x + threadIdx.x) >> 5;  // P_*N_
  const int lane = threadIdx.x & 31;
  const int i = wid % N_;
  const int p = wid / N_;

  float sv[NHEADS], mh[NHEADS], sh[NHEADS];
#pragma unroll
  for (int hd = 0; hd < NHEADS; ++hd) {
    sv[hd] = s_src[(size_t)(p * NHEADS + hd) * N_ + i];
    mh[hd] = -3.0e38f;
    sh[hd] = 0.f;
  }
  const float* arow = adjs + ((size_t)p * N_ + i) * N_;
  const float* sdb  = s_dst + (size_t)p * NHEADS * N_;
  for (int j = lane; j < N_; j += 32) {
    if (j + 256 < N_) __builtin_prefetch(arow + j + 256, 0, 0);
    const float a = arow[j];
    if (a > 0.f) {
#pragma unroll
      for (int hd = 0; hd < NHEADS; ++hd) {
        float e  = sv[hd] + sdb[hd * N_ + j];
        e        = e > 0.f ? e : ALPHA * e;
        float nm = fmaxf(mh[hd], e);
        sh[hd]   = sh[hd] * __expf(mh[hd] - nm) + __expf(e - nm);
        mh[hd]   = nm;
      }
    }
  }
#pragma unroll
  for (int hd = 0; hd < NHEADS; ++hd) {
#pragma unroll
    for (int off = 16; off >= 1; off >>= 1) {
      float om = __shfl_xor(mh[hd], off, 32);
      float os = __shfl_xor(sh[hd], off, 32);
      float nm = fmaxf(mh[hd], om);
      sh[hd]   = sh[hd] * __expf(mh[hd] - nm) + os * __expf(om - nm);
      mh[hd]   = nm;
    }
    if (lane == 0) {
      rmax[(size_t)(p * NHEADS + hd) * N_ + i] = mh[hd];
      rinv[(size_t)(p * NHEADS + hd) * N_ + i] =
          (sh[hd] > 0.f) ? 1.f / sh[hd] : 0.f;
    }
  }
}

// ---------------------------------------------------------------------------
// Fused attention GEMM: block = (p, 16-row tile), 4 waves = 4 heads.
// adj 16x32 tile staged by async-to-LDS (triple-buffered, 1 barrier/K-step),
// shared by all heads. Each wave: regen A from LDS -> 8 WMMAs -> ELU -> m.
// ---------------------------------------------------------------------------
__global__ __launch_bounds__(128) void att_gemm_kernel(
    const float* __restrict__ adjs, const float* __restrict__ s_src,
    const float* __restrict__ s_dst, const float* __restrict__ rmax,
    const float* __restrict__ rinv, const _Float16* __restrict__ hT,
    float* __restrict__ m_out) {
  __shared__ __align__(16) float adjS[3][16 * 32];      // 3 x 2KB
  const int tid  = threadIdx.x;           // 128 threads
  const int hd   = tid >> 5;              // wave = head
  const int lane = tid & 31;
  const int p       = blockIdx.x / (N_ / 16);
  const int rowbase = (blockIdx.x % (N_ / 16)) * 16;
  const int ph = p * NHEADS + hd;
  const int i  = rowbase + (lane & 15);
  const int kc = (lane >> 4) * 16;

  const float sv = s_src[(size_t)ph * N_ + i];
  const float rm = rmax[(size_t)ph * N_ + i];
  const float ri = rinv[(size_t)ph * N_ + i];
  const float* sdv = s_dst + (size_t)ph * N_;

  // async adj loader mapping: 128 threads -> 16 rows x 8 float4 (16B each)
  const int lr = tid >> 3;                // 0..15
  const int lc = (tid & 7) * 4;           // 0,4,...,28
  const float* garow = adjs + ((size_t)p * N_ + rowbase + lr) * N_ + lc;

  v8f acc[8];
#pragma unroll
  for (int dt = 0; dt < 8; ++dt)
#pragma unroll
    for (int z = 0; z < 8; ++z) acc[dt][z] = 0.f;

  constexpr int KT = N_ / 32;             // 64 K-steps
  async_ld_b128(lds_off_u32(&adjS[0][lr * 32 + lc]), garow);   // tile 0

  for (int kt = 0; kt < KT; ++kt) {
    const int k0 = kt * 32;
    if (kt + 1 < KT) {
      async_ld_b128(lds_off_u32(&adjS[(kt + 1) % 3][lr * 32 + lc]),
                    garow + (kt + 1) * 32);
      wait_async1();
    } else {
      wait_async0();
    }
    __syncthreads();
    const float* abuf = &adjS[kt % 3][0];

    v16h A;
#pragma unroll
    for (int c = 0; c < 2; ++c) {
      const int kl = kc + c * 8;                          // local K base
      const float4 m0 = *(const float4*)(abuf + (lane & 15) * 32 + kl);
      const float4 m1 = *(const float4*)(abuf + (lane & 15) * 32 + kl + 4);
      const float4 d0 = *(const float4*)(sdv + k0 + kl);
      const float4 d1 = *(const float4*)(sdv + k0 + kl + 4);
      const float mv[8] = {m0.x, m0.y, m0.z, m0.w, m1.x, m1.y, m1.z, m1.w};
      const float dv[8] = {d0.x, d0.y, d0.z, d0.w, d1.x, d1.y, d1.z, d1.w};
#pragma unroll
      for (int t = 0; t < 8; ++t) {
        float e = sv + dv[t];
        e = e > 0.f ? e : ALPHA * e;
        const float a = (mv[t] > 0.f) ? __expf(e - rm) * ri : 0.f;
        A[c * 8 + t] = (_Float16)a;
      }
    }
    v16h Bt[8];
#pragma unroll
    for (int dt = 0; dt < 8; ++dt) {
      const int d = dt * 16 + (lane & 15);
      Bt[dt] = *(const v16h*)(hT + ((size_t)ph * NHID + d) * N_ + k0 +
                              (lane >> 4) * 16);
    }
#pragma unroll
    for (int dt = 0; dt < 8; ++dt) acc[dt] = wmma_f16(A, Bt[dt], acc[dt]);
  }
  // ELU epilogue, write m[p][node][hd*NHID + d]
  const int nodebase = rowbase + 8 * (lane >> 4);
#pragma unroll
  for (int dt = 0; dt < 8; ++dt) {
    const int col = hd * NHID + dt * 16 + (lane & 15);
#pragma unroll
    for (int r = 0; r < 8; ++r) {
      float v = acc[dt][r];
      v = v > 0.f ? v : expm1f(v);
      m_out[((size_t)p * N_ + (nodebase + r)) * KF + col] = v;
    }
  }
}

// ---------------------------------------------------------------------------
// scores init (graph replays must re-zero; atomics accumulate below)
// ---------------------------------------------------------------------------
__global__ void init_scores_kernel(float* __restrict__ scores) {
  if (threadIdx.x < P_) scores[threadIdx.x] = 0.f;
}

// ---------------------------------------------------------------------------
// Semantic attention: hs = tanh(m @ Wsem + b); scores[p] += sum_n hs.q
// WMMA GEMM (6144x512)@(512x128), 16 rows/wave, 8 waves/block, 48 blocks.
// ---------------------------------------------------------------------------
__global__ __launch_bounds__(256) void semantic_kernel(
    const float* __restrict__ m_in, const _Float16* __restrict__ WsemT,
    const float* __restrict__ b_sem, const float* __restrict__ q_sem,
    float* __restrict__ scores) {
  const int wave = threadIdx.x >> 5;
  const int lane = threadIdx.x & 31;
  const int rowbase = blockIdx.x * 128 + wave * 16;     // in [0, P_*N_)
  const int row     = rowbase + (lane & 15);
  const int kc      = (lane >> 4) * 16;
  const float* arow = m_in + (size_t)row * KF;

  v8f acc[8];
#pragma unroll
  for (int dt = 0; dt < 8; ++dt)
#pragma unroll
    for (int z = 0; z < 8; ++z) acc[dt][z] = 0.f;

  for (int k0 = 0; k0 < KF; k0 += 32) {
    v16h A;
#pragma unroll
    for (int c = 0; c < 2; ++c) {
      const int kb = k0 + kc + c * 8;
      const float4 a0 = *(const float4*)(arow + kb);
      const float4 a1 = *(const float4*)(arow + kb + 4);
      const float av[8] = {a0.x, a0.y, a0.z, a0.w, a1.x, a1.y, a1.z, a1.w};
#pragma unroll
      for (int t = 0; t < 8; ++t) A[c * 8 + t] = (_Float16)av[t];
    }
    v16h Bt[8];
#pragma unroll
    for (int dt = 0; dt < 8; ++dt) {
      const int scol = dt * 16 + (lane & 15);
      const _Float16* brow = WsemT + (size_t)scol * KF;
      Bt[dt] = *(const v16h*)(brow + k0 + (lane >> 4) * 16);
    }
#pragma unroll
    for (int dt = 0; dt < 8; ++dt) acc[dt] = wmma_f16(A, Bt[dt], acc[dt]);
  }
  // tanh + dot with q, summed over this lane's cols and 8 rows
  float tot = 0.f;
#pragma unroll
  for (int dt = 0; dt < 8; ++dt) {
    const int scol = dt * 16 + (lane & 15);
    const float b = b_sem[scol];
    const float q = q_sem[scol];
#pragma unroll
    for (int r = 0; r < 8; ++r) tot += tanhf(acc[dt][r] + b) * q;
  }
  // reduce over the 16 col-lanes within each half-wave
#pragma unroll
  for (int off = 1; off < 16; off <<= 1) tot += __shfl_xor(tot, off, 32);
  if (lane == 0 || lane == 16) {
    atomicAdd(&scores[rowbase / N_], tot);   // 16-row tile never crosses p
  }
}

// ---------------------------------------------------------------------------
// final: w = softmax(scores/N); out = sum_p w[p] * m[p]
// ---------------------------------------------------------------------------
__global__ void combine_kernel(const float* __restrict__ m_in,
                               const float* __restrict__ scores,
                               float* __restrict__ out) {
  const int i = blockIdx.x * blockDim.x + threadIdx.x;   // N_*KF
  const float s0 = scores[0] * (1.f / N_);
  const float s1 = scores[1] * (1.f / N_);
  const float s2 = scores[2] * (1.f / N_);
  const float mx = fmaxf(s0, fmaxf(s1, s2));
  const float w0 = __expf(s0 - mx), w1 = __expf(s1 - mx), w2 = __expf(s2 - mx);
  const float inv = 1.f / (w0 + w1 + w2);
  const size_t stride = (size_t)N_ * KF;
  out[i] = (m_in[i] * w0 + m_in[i + stride] * w1 + m_in[i + 2 * stride] * w2) * inv;
}

// ---------------------------------------------------------------------------
extern "C" void kernel_launch(void* const* d_in, const int* in_sizes, int n_in,
                              void* d_out, int out_size, void* d_ws,
                              size_t ws_size, hipStream_t stream) {
  const float* x      = (const float*)d_in[0];
  const float* adjs   = (const float*)d_in[1];
  const float* W_node = (const float*)d_in[2];
  const float* a_node = (const float*)d_in[3];
  const float* W_sem  = (const float*)d_in[4];
  const float* b_sem  = (const float*)d_in[5];
  const float* q_sem  = (const float*)d_in[6];
  float* out = (float*)d_out;

  char* base = (char*)d_ws;
  size_t off = 0;
  auto alloc = [&](size_t bytes) -> void* {
    void* pp = base + off;
    off = (off + bytes + 255) & ~(size_t)255;
    return pp;
  };
  _Float16* xh    = (_Float16*)alloc(sizeof(_Float16) * (size_t)N_ * NFEAT);
  _Float16* WT    = (_Float16*)alloc(sizeof(_Float16) * (size_t)PH * NHID * NFEAT);
  _Float16* hT    = (_Float16*)alloc(sizeof(_Float16) * (size_t)PH * NHID * N_);
  _Float16* WsemT = (_Float16*)alloc(sizeof(_Float16) * (size_t)SHID * KF);
  float* s_src = (float*)alloc(sizeof(float) * (size_t)PH * N_);
  float* s_dst = (float*)alloc(sizeof(float) * (size_t)PH * N_);
  float* rmx   = (float*)alloc(sizeof(float) * (size_t)PH * N_);
  float* rin   = (float*)alloc(sizeof(float) * (size_t)PH * N_);
  float* m_ws  = (float*)alloc(sizeof(float) * (size_t)P_ * N_ * KF);
  float* scr   = (float*)alloc(sizeof(float) * 16);

  cvt_f32_to_f16_kernel<<<(N_ * NFEAT) / 256, 256, 0, stream>>>(x, xh);
  wnodeT_kernel<<<(PH * NHID * NFEAT) / 256, 256, 0, stream>>>(W_node, WT);
  wsemT_kernel<<<(SHID * KF) / 256, 256, 0, stream>>>(W_sem, WsemT);

  gemm_h_kernel<<<PH * (N_ / 128), 256, 0, stream>>>(xh, WT, hT);
  node_scores_kernel<<<(PH * N_) / 256, 256, 0, stream>>>(hT, a_node, s_src, s_dst);
  stats_kernel<<<(P_ * N_) / 8, 256, 0, stream>>>(adjs, s_src, s_dst, rmx, rin);
  att_gemm_kernel<<<P_ * (N_ / 16), 128, 0, stream>>>(adjs, s_src, s_dst, rmx,
                                                      rin, hT, m_ws);
  init_scores_kernel<<<1, 32, 0, stream>>>(scr);
  semantic_kernel<<<(P_ * N_) / 128, 256, 0, stream>>>(m_ws, WsemT, b_sem,
                                                       q_sem, scr);
  combine_kernel<<<((size_t)N_ * KF) / 256, 256, 0, stream>>>(m_ws, scr, out);
}
